// DorsalPathway0_11716670784093
// MI455X (gfx1250) — matrix operand order repaired and checked by reading, not compiled, not run on hardware
//
#include <hip/hip_runtime.h>
#include <hip/hip_bf16.h>

// Problem dims
#define B_   256
#define S_   512
#define NIN  2048
#define NOUT 512
#define H_   30
#define HP   32      // hidden padded to 32 (WMMA K)
#define G4   120     // 4*H
#define GP   128     // gates padded to 128 (8 N-tiles)
#define M_   (B_*S_) // 131072 flattened rows
#define BN_EPS 1e-5f

typedef __attribute__((ext_vector_type(16))) __bf16 bf16x16;
typedef __attribute__((ext_vector_type(8)))  float  f32x8;
typedef __attribute__((ext_vector_type(4)))  unsigned int u32x4;
typedef __attribute__((ext_vector_type(4)))  int i32x4;
typedef __attribute__((ext_vector_type(8)))  int i32x8;

__device__ __forceinline__ f32x8 wmma_bf16(bf16x16 a, bf16x16 b, f32x8 c) {
  // D = A(16x32 bf16) * B(32x16 bf16) + C(16x16 f32)
  return __builtin_amdgcn_wmma_f32_16x16x32_bf16(false, a, false, b, (short)0, c, false, false);
}

// A-matrix 16-bit layout (contiguous, matching B): lanes 0-15 hold K=0..15,
// lanes 16-31 hold K=16..31; element e of the v16 maps to:
__device__ __forceinline__ int a_koff(int e, int hl) {
  return e + hl * 16;
}

__device__ __forceinline__ float sigm(float x) { return 1.0f / (1.0f + __expf(-x)); }

// ---------------------------------------------------------------------------
// Tensor Data Mover: DMA a 2-D tile of f32/dword data (rows x row_dwords,
// global row stride stride_dwords) into LDS.  Descriptor per ISA ch.8:
//  group0: [1:0]count=1, [63:32]lds_addr, [120:64]global_addr, [127:126]type=2
//  group1: [17:16]data_size=2 (4B), [79:48]tensor_dim0, [111:80]tensor_dim1,
//          [127:112]tile_dim0, [143:128]tile_dim1, [207:160]tensor_dim0_stride
// Issued by one wave (EXEC ignored by TDM); completion via TENSORcnt.
// ---------------------------------------------------------------------------
__device__ __forceinline__ void tdm_load_2d_to_lds(void* lds_dst, const void* gsrc,
                                                   unsigned row_dwords, unsigned rows,
                                                   unsigned stride_dwords) {
  unsigned lds = (unsigned)(size_t)lds_dst;                 // LDS aperture: low 32b
  unsigned long long ga = (unsigned long long)(size_t)gsrc; // 57b global address
  u32x4 g0;
  g0[0] = 1u;                                                // count=1 user D#
  g0[1] = lds;                                               // lds_addr
  g0[2] = (unsigned)(ga & 0xffffffffu);                      // global_addr[31:0]
  g0[3] = (unsigned)((ga >> 32) & 0x1ffffffu) | (2u << 30);  // ga[56:32] | type=2
  i32x8 g1;
  g1[0] = (int)(2u << 16);            // data_size = 4 bytes
  g1[1] = (int)(row_dwords << 16);    // tensor_dim0[15:0]  (bits 63:48)
  g1[2] = (int)(rows << 16);          // tensor_dim0[31:16]=0 | tensor_dim1[15:0]
  g1[3] = (int)(row_dwords << 16);    // tensor_dim1[31:16]=0 | tile_dim0
  g1[4] = (int)rows;                  // tile_dim1 | tile_dim2=0
  g1[5] = (int)stride_dwords;         // tensor_dim0_stride[31:0]
  g1[6] = 0;                          // stride[47:32] | tensor_dim1_stride[15:0]
  g1[7] = 0;                          // tensor_dim1_stride[47:16]
  i32x4 z4 = {0, 0, 0, 0};
#if __has_include(<hip/amd_detail/amd_gfx1250_TDM.h>)
  i32x8 z8 = {0, 0, 0, 0, 0, 0, 0, 0};
  __builtin_amdgcn_tensor_load_to_lds(g0, g1, z4, z4, z8, 0);   // clang-23 lane
#else
  __builtin_amdgcn_tensor_load_to_lds(g0, g1, z4, z4, 0);       // ROCm 7.2 lane
#endif
}

// ---------------------------------------------------------------------------
// zero BN stat accumulators (ws is poisoned; must re-zero every launch)
// ---------------------------------------------------------------------------
__global__ void zero_stats(float* a, float* b) {
  int t = threadIdx.x;
  if (t < HP) { a[t] = 0.0f; b[t] = 0.0f; }
}

// ---------------------------------------------------------------------------
// prep_w1: pad+transpose W1 [2048,30] f32 -> W1pT [32,2048] bf16 so the fc1
// K-loop has branch-free, contiguous B-fragment loads. Grid 256x256.
// ---------------------------------------------------------------------------
__global__ void __launch_bounds__(256)
prep_w1(const float* __restrict__ W1, __bf16* __restrict__ W1pT) {
  int idx = blockIdx.x * 256 + threadIdx.x;   // 32*2048 total
  int n = idx >> 11, k = idx & (NIN - 1);
  W1pT[idx] = (n < H_) ? (__bf16)W1[k * H_ + n] : (__bf16)0.0f;
}

// ---------------------------------------------------------------------------
// fc1: flat = relu(x @ W1 + b1), store bf16 padded [M_,32]; accumulate
// per-column sum / sumsq for BatchNorm via LDS + global float atomics.
// Streaming GEMM: x read once (1.07 GB) -> HBM-bound; plain b128 loads.
// ---------------------------------------------------------------------------
__global__ void __launch_bounds__(256)
fc1_kernel(const float* __restrict__ x, const __bf16* __restrict__ W1pT,
           const float* __restrict__ b1v, __bf16* __restrict__ flat,
           float* __restrict__ colsum, float* __restrict__ colsumsq) {
  __shared__ float lsum[HP], lsumsq[HP];
  const int tid = threadIdx.x;
  if (tid < HP) { lsum[tid] = 0.0f; lsumsq[tid] = 0.0f; }
  __syncthreads();

  const int lane = tid & 31, wv = tid >> 5;
  const int hl = lane >> 4, ln = lane & 15;
  const int mA = blockIdx.x * 128 + wv * 16 + ln;        // A row this lane feeds
  const float* xrow = x + (size_t)mA * NIN;

  f32x8 acc0 = {}; f32x8 acc1 = {};
  const int koff = hl * 16;
  const int n0 = ln, n1 = 16 + ln;
  const __bf16* wrow0 = W1pT + (size_t)n0 * NIN;
  const __bf16* wrow1 = W1pT + (size_t)n1 * NIN;

  for (int kb = 0; kb < NIN; kb += 32) {
    if (kb + 64 < NIN) __builtin_prefetch(xrow + kb + 64, 0, 1);  // global_prefetch_b8
    bf16x16 a, b0, bx1;
#pragma unroll
    for (int e = 0; e < 16; ++e) a[e] = (__bf16)xrow[kb + a_koff(e, hl)];
#pragma unroll
    for (int e = 0; e < 16; ++e) {
      b0[e]  = wrow0[kb + koff + e];   // contiguous, zero-padded, branch-free
      bx1[e] = wrow1[kb + koff + e];
    }
    acc0 = wmma_bf16(a, b0, acc0);
    acc1 = wmma_bf16(a, bx1, acc1);
  }

  const int mbase = blockIdx.x * 128 + wv * 16 + hl * 8; // D rows = r + 8*hl
  { // N-tile 0 (cols 0..15, all valid)
    float bias = b1v[n0];
    float s = 0.0f, sq = 0.0f;
#pragma unroll
    for (int r = 0; r < 8; ++r) {
      float v = fmaxf(acc0[r] + bias, 0.0f);
      flat[(size_t)(mbase + r) * HP + n0] = (__bf16)v;
      s += v; sq += v * v;
    }
    atomicAdd(&lsum[n0], s); atomicAdd(&lsumsq[n0], sq);
  }
  { // N-tile 1 (cols 16..31; pad cols have acc=0,bias=0 -> store 0)
    float bias = (n1 < H_) ? b1v[n1] : 0.0f;
    float s = 0.0f, sq = 0.0f;
#pragma unroll
    for (int r = 0; r < 8; ++r) {
      float v = fmaxf(acc1[r] + bias, 0.0f);
      flat[(size_t)(mbase + r) * HP + n1] = (__bf16)v;
      s += v; sq += v * v;
    }
    if (n1 < H_) { atomicAdd(&lsum[n1], s); atomicAdd(&lsumsq[n1], sq); }
  }
  __syncthreads();
  if (tid < H_) {
    atomicAdd(&colsum[tid], lsum[tid]);
    atomicAdd(&colsumsq[tid], lsumsq[tid]);
  }
}

// ---------------------------------------------------------------------------
// prep: finalize BN stats, fold BN into layer-0 input weights, convert all
// LSTM / output weights to padded bf16 layouts. 1 block x 128 threads.
// ---------------------------------------------------------------------------
__global__ void __launch_bounds__(128)
prep_kernel(const float* __restrict__ colsum, const float* __restrict__ colsumsq,
            const float* __restrict__ Wih0, const float* __restrict__ Whh0,
            const float* __restrict__ bias0,
            const float* __restrict__ Wih1, const float* __restrict__ Whh1,
            const float* __restrict__ bias1, const float* __restrict__ Wout,
            __bf16* __restrict__ Wih0p, __bf16* __restrict__ Whh0p,
            __bf16* __restrict__ Wih1p, __bf16* __restrict__ Whh1p,
            float* __restrict__ bias0p, float* __restrict__ bias1p,
            __bf16* __restrict__ Woutp) {
  __shared__ float mus[HP], rstds[HP];
  const int tid = threadIdx.x;
  if (tid < HP) {
    if (tid < H_) {
      float inv = 1.0f / (float)M_;
      float mu = colsum[tid] * inv;
      float var = colsumsq[tid] * inv - mu * mu;
      mus[tid] = mu; rstds[tid] = rsqrtf(var + BN_EPS);
    } else { mus[tid] = 0.0f; rstds[tid] = 0.0f; }  // pad columns contribute 0
  }
  __syncthreads();

  const int j = tid;  // 0..127 (gate row)
  float bcorr = 0.0f;
  for (int k = 0; k < HP; ++k) {
    bool v = (j < G4) && (k < H_);
    float wih0 = v ? Wih0[j * H_ + k] : 0.0f;
    float whh0 = v ? Whh0[j * H_ + k] : 0.0f;
    float wih1 = v ? Wih1[j * H_ + k] : 0.0f;
    float whh1 = v ? Whh1[j * H_ + k] : 0.0f;
    float w0s = wih0 * rstds[k];
    Wih0p[j * HP + k] = (__bf16)w0s;
    Whh0p[j * HP + k] = (__bf16)whh0;
    Wih1p[j * HP + k] = (__bf16)wih1;
    Whh1p[j * HP + k] = (__bf16)whh1;
    bcorr += mus[k] * w0s;
  }
  bias0p[j] = (j < G4 ? bias0[j] : 0.0f) - bcorr;
  bias1p[j] = (j < G4 ? bias1[j] : 0.0f);

  for (int idx = tid; idx < HP * NOUT; idx += 128) {
    int k = idx >> 9, n = idx & (NOUT - 1);
    Woutp[idx] = (k < H_) ? (__bf16)Wout[k * NOUT + n] : (__bf16)0.0f;
  }
}

// ---------------------------------------------------------------------------
// proj: X = in @ Wp^T + biasp  (whole-layer input projections, hoisted out of
// the recurrence). Weights staged into LDS by one TDM descriptor (8 KB).
// remap=1: output row b*512+s -> s*256+b (time-major for the recurrence).
// ---------------------------------------------------------------------------
__global__ void __launch_bounds__(256)
proj_kernel(const __bf16* __restrict__ in, const __bf16* __restrict__ Wp,
            const float* __restrict__ biasp, float* __restrict__ X, int remap) {
  __shared__ __bf16 w[GP * HP];  // 8 KB
  const int tid = threadIdx.x;
  if (tid < 32) {                          // wave 0 drives the DMA
    tdm_load_2d_to_lds(w, Wp, (GP * HP * 2) / 4, 1, (GP * HP * 2) / 4);
    __builtin_amdgcn_s_wait_tensorcnt(0);
  }
  __syncthreads();

  const int lane = tid & 31, wv = tid >> 5;
  const int hl = lane >> 4, ln = lane & 15;
  const int mA = blockIdx.x * 128 + wv * 16 + ln;
  const __bf16* inrow = in + (size_t)mA * HP;
  bf16x16 a;
#pragma unroll
  for (int e = 0; e < 16; ++e) a[e] = inrow[a_koff(e, hl)];

  const int koff = hl * 16;
  const int mbase = blockIdx.x * 128 + wv * 16 + hl * 8;
#pragma unroll
  for (int nt = 0; nt < 8; ++nt) {
    int n = nt * 16 + ln;
    bf16x16 b;
#pragma unroll
    for (int e = 0; e < 16; ++e) b[e] = w[n * HP + koff + e];  // B[k][n] = W[n][k]
    float bias = biasp[n];
    f32x8 acc;
#pragma unroll
    for (int r = 0; r < 8; ++r) acc[r] = bias;
    acc = wmma_bf16(a, b, acc);
#pragma unroll
    for (int r = 0; r < 8; ++r) {
      int mo = mbase + r;
      int mrow = remap ? ((mo & (S_ - 1)) * B_ + (mo >> 9)) : mo;
      X[(size_t)mrow * GP + n] = acc[r];
    }
  }
}

// ---------------------------------------------------------------------------
// LSTM recurrence: single persistent workgroup (16 waves, 512 threads).
// X[s] tiles (128 KB) double-buffered in LDS via TDM: wave 0 issues the DMA
// for X[s+1] at the top of step s, overlapping it with the step's 8 v_wmma
// per wave + gate math; s_wait_tensorcnt + barrier close the dependency.
// z aliases the consumed X buffer (each wave touches only its own 16 rows).
// Dynamic LDS = 2*128K (X/z) + 16K (h) + 8K (Whh) = 280 KB (< 320 KB/WGP).
// ---------------------------------------------------------------------------
__global__ void __launch_bounds__(512)
lstm_kernel(const float* __restrict__ X, const __bf16* __restrict__ Whhp,
            const float* __restrict__ h0, const float* __restrict__ c0,
            __bf16* __restrict__ ybf, float* __restrict__ yf32) {
  extern __shared__ char smem[];
  const size_t XBYTES = (size_t)B_ * GP * sizeof(float);     // 128 KB
  float*  xb0  = (float*)smem;
  float*  xb1  = (float*)(smem + XBYTES);
  __bf16* hbuf = (__bf16*)(smem + 2 * XBYTES);               // [256][32]
  __bf16* whh  = hbuf + B_ * HP;                             // [128][32]

  const int tid = threadIdx.x, lane = tid & 31, wv = tid >> 5;
  const int hl = lane >> 4, ln = lane & 15;

  if (tid < 32) {                       // wave 0: DMA Whh (8 KB) and X[0] (128 KB)
    tdm_load_2d_to_lds(whh, Whhp, (GP * HP * 2) / 4, 1, (GP * HP * 2) / 4);
    tdm_load_2d_to_lds(xb0, X, GP, B_, GP);
  }
  for (int i = tid; i < B_ * HP; i += 512) {
    int row = i >> 5, u = i & 31;
    hbuf[i] = (u < H_) ? (__bf16)h0[row * H_ + u] : (__bf16)0.0f;
  }
  float creg[16];
#pragma unroll
  for (int j = 0; j < 16; ++j) {
    int idx = tid + j * 512, row = idx >> 5, u = idx & 31;
    creg[j] = (u < H_) ? c0[row * H_ + u] : 0.0f;
  }
  if (tid < 32) __builtin_amdgcn_s_wait_tensorcnt(0);
  __syncthreads();

  const int koff = hl * 16;
  const int mbase = wv * 16 + hl * 8;
  const __bf16* hrow = hbuf + (size_t)(wv * 16 + ln) * HP;

  for (int s = 0; s < S_; ++s) {
    float* xcur = (s & 1) ? xb1 : xb0;
    float* xnxt = (s & 1) ? xb0 : xb1;
    if (tid < 32 && s + 1 < S_)         // prefetch next timestep's projections
      tdm_load_2d_to_lds(xnxt, X + (size_t)(s + 1) * B_ * GP, GP, B_, GP);

    // z = X[s] + h @ Whh^T  (z overwrites the consumed X tile in LDS)
    bf16x16 a;
#pragma unroll
    for (int e = 0; e < 16; ++e) a[e] = hrow[a_koff(e, hl)];
#pragma unroll
    for (int nt = 0; nt < 8; ++nt) {
      int n = nt * 16 + ln;
      bf16x16 b;
#pragma unroll
      for (int e = 0; e < 16; ++e) b[e] = whh[n * HP + koff + e];
      f32x8 acc;
#pragma unroll
      for (int r = 0; r < 8; ++r) acc[r] = xcur[(mbase + r) * GP + n];
      acc = wmma_bf16(a, b, acc);
#pragma unroll
      for (int r = 0; r < 8; ++r) xcur[(mbase + r) * GP + n] = acc[r];
    }
    __syncthreads();

    // elementwise gates: i,f,g,o at columns u, 30+u, 60+u, 90+u
#pragma unroll
    for (int j = 0; j < 16; ++j) {
      int idx = tid + j * 512, row = idx >> 5, u = idx & 31;
      if (u < H_) {
        float zi = xcur[row * GP + u];
        float zf = xcur[row * GP + H_ + u];
        float zg = xcur[row * GP + 2 * H_ + u];
        float zo = xcur[row * GP + 3 * H_ + u];
        float c = sigm(zf) * creg[j] + sigm(zi) * tanhf(zg);
        float h = sigm(zo) * tanhf(c);
        creg[j] = c;
        hbuf[row * HP + u] = (__bf16)h;
        ybf[((size_t)s * B_ + row) * HP + u] = (__bf16)h;
        if (yf32) yf32[((size_t)row * S_ + s) * H_ + u] = h;
      } else {
        hbuf[row * HP + u] = (__bf16)0.0f;
        ybf[((size_t)s * B_ + row) * HP + u] = (__bf16)0.0f;
      }
    }
    if (tid < 32) __builtin_amdgcn_s_wait_tensorcnt(0);  // X[s+1] landed
    __syncthreads();
  }
}

// ---------------------------------------------------------------------------
// out: dorsal = y1 @ Wout + bout.  y1: [M_,32] bf16 time-major; Wout staged
// into 32 KB LDS by one TDM descriptor; single K chunk, 32 N-tiles per wave.
// Write-bound (268 MB out) -> minimal overhead per store.
// ---------------------------------------------------------------------------
__global__ void __launch_bounds__(256)
out_kernel(const __bf16* __restrict__ y1, const __bf16* __restrict__ Woutp,
           const float* __restrict__ bout, float* __restrict__ out) {
  __shared__ __bf16 w[HP * NOUT];  // 32 KB
  const int tid = threadIdx.x;
  if (tid < 32) {
    tdm_load_2d_to_lds(w, Woutp, (HP * NOUT * 2) / 4, 1, (HP * NOUT * 2) / 4);
    __builtin_amdgcn_s_wait_tensorcnt(0);
  }
  __syncthreads();

  const int lane = tid & 31, wv = tid >> 5;
  const int hl = lane >> 4, ln = lane & 15;
  const int mA = blockIdx.x * 128 + wv * 16 + ln;          // output row (b-major)
  const int rA = (mA & (S_ - 1)) * B_ + (mA >> 9);         // y1 row (time-major)
  const __bf16* yrow = y1 + (size_t)rA * HP;
  bf16x16 a;
#pragma unroll
  for (int e = 0; e < 16; ++e) a[e] = yrow[a_koff(e, hl)];

  const int koff = hl * 16;
  const int mbase = blockIdx.x * 128 + wv * 16 + hl * 8;
  for (int nt = 0; nt < 32; ++nt) {
    int n = nt * 16 + ln;
    bf16x16 b;
#pragma unroll
    for (int e = 0; e < 16; ++e) b[e] = w[(koff + e) * NOUT + n];  // [k][n]
    float bb = bout[n];
    f32x8 acc;
#pragma unroll
    for (int r = 0; r < 8; ++r) acc[r] = bb;
    acc = wmma_bf16(a, b, acc);
#pragma unroll
    for (int r = 0; r < 8; ++r) out[(size_t)(mbase + r) * NOUT + n] = acc[r];
  }
}

// ---------------------------------------------------------------------------
extern "C" void kernel_launch(void* const* d_in, const int* in_sizes, int n_in,
                              void* d_out, int out_size, void* d_ws, size_t ws_size,
                              hipStream_t stream) {
  const float* x     = (const float*)d_in[0];
  const float* h0    = (const float*)d_in[1];
  const float* c0    = (const float*)d_in[2];
  const float* W1    = (const float*)d_in[3];
  const float* b1    = (const float*)d_in[4];
  const float* Wih0  = (const float*)d_in[5];
  const float* Whh0  = (const float*)d_in[6];
  const float* bias0 = (const float*)d_in[7];
  const float* Wih1  = (const float*)d_in[8];
  const float* Whh1  = (const float*)d_in[9];
  const float* bias1 = (const float*)d_in[10];
  const float* Wout  = (const float*)d_in[11];
  const float* bout  = (const float*)d_in[12];

  char* ws = (char*)d_ws;
  size_t off = 0;
  auto alloc = [&](size_t bytes) -> char* {
    char* p = ws + off;
    off += (bytes + 255) & ~(size_t)255;
    return p;
  };
  float*  colsum   = (float*)alloc(HP * sizeof(float));
  float*  colsumsq = (float*)alloc(HP * sizeof(float));
  float*  bias0p   = (float*)alloc(GP * sizeof(float));
  float*  bias1p   = (float*)alloc(GP * sizeof(float));
  __bf16* W1pT     = (__bf16*)alloc((size_t)HP * NIN * 2);  // 128 KB
  __bf16* Wih0p    = (__bf16*)alloc(GP * HP * 2);
  __bf16* Whh0p    = (__bf16*)alloc(GP * HP * 2);
  __bf16* Wih1p    = (__bf16*)alloc(GP * HP * 2);
  __bf16* Whh1p    = (__bf16*)alloc(GP * HP * 2);
  __bf16* Woutp    = (__bf16*)alloc(HP * NOUT * 2);
  __bf16* flatbuf  = (__bf16*)alloc((size_t)M_ * HP * 2);   // 8 MB
  __bf16* y0buf    = (__bf16*)alloc((size_t)M_ * HP * 2);   // 8 MB
  float*  Xbuf     = (float*)alloc((size_t)M_ * GP * 4);    // 64 MB

  float* dorsal = (float*)d_out;
  float* rnnout = dorsal + (size_t)M_ * NOUT;
  __bf16* y1buf = flatbuf;  // flat is dead after layer-0 projection -> reuse

  const size_t lstm_lds = 2 * (size_t)B_ * GP * 4 + (size_t)B_ * HP * 2 + (size_t)GP * HP * 2;

  zero_stats<<<1, 64, 0, stream>>>(colsum, colsumsq);
  prep_w1<<<(HP * NIN) / 256, 256, 0, stream>>>(W1, W1pT);
  fc1_kernel<<<M_ / 128, 256, 0, stream>>>(x, W1pT, b1, flatbuf, colsum, colsumsq);
  prep_kernel<<<1, 128, 0, stream>>>(colsum, colsumsq, Wih0, Whh0, bias0,
                                     Wih1, Whh1, bias1, Wout,
                                     Wih0p, Whh0p, Wih1p, Whh1p,
                                     bias0p, bias1p, Woutp);
  // layer 0: hoisted input projection (BN folded in), then recurrence
  proj_kernel<<<M_ / 128, 256, 0, stream>>>(flatbuf, Wih0p, bias0p, Xbuf, 1);
  lstm_kernel<<<1, 512, lstm_lds, stream>>>(Xbuf, Whh0p, h0, c0, y0buf, nullptr);
  // layer 1
  proj_kernel<<<M_ / 128, 256, 0, stream>>>(y0buf, Wih1p, bias1p, Xbuf, 0);
  lstm_kernel<<<1, 512, lstm_lds, stream>>>(Xbuf, Whh1p, h0 + B_ * H_, c0 + B_ * H_,
                                            y1buf, rnnout);
  // output projection
  out_kernel<<<M_ / 128, 256, 0, stream>>>(y1buf, Woutp, bout, dorsal);
}
